// TransformerModel_72155450573151
// MI455X (gfx1250) — matrix operand order repaired and checked by reading, not compiled, not run on hardware
//
#include <hip/hip_runtime.h>

// ---------------- model constants ----------------
#define D_    512
#define H_    8
#define HD_   64
#define DFF_  2048
#define V_    32000
#define B_    4
#define S_    512
#define T_    512
#define LE_   6
#define LD_   6
#define M_    (B_ * S_)          // 2048 rows for both encoder and decoder streams
#define EPS_  1e-5f
#define SCALE_ 0.125f            // 1/sqrt(64)
#define SQRTD_ 22.62741699796952f

typedef __bf16 bf16_t;
typedef __attribute__((ext_vector_type(16))) __bf16 v16bf;
typedef __attribute__((ext_vector_type(8)))  float  v8f;

// ---------------- CDNA5 async Global->LDS copy (ASYNCcnt-tracked) ----------------
// 16B per lane: copies global vaddr to LDS at vdst. LDS byte offset = low 32 bits of the
// generic shared-aperture pointer.
static __device__ __forceinline__ void async_cp_b128(bf16_t* l, const bf16_t* g) {
  unsigned lds32 = (unsigned)(unsigned long long)l;
  unsigned long long ga = (unsigned long long)g;
  asm volatile("global_load_async_to_lds_b128 %0, %1, off"
               :: "v"(lds32), "v"(ga) : "memory");
}
template <int N>
static __device__ __forceinline__ void wait_asyncN() {
  asm volatile("s_wait_asynccnt %0" :: "i"(N) : "memory");
}

// ---------------- WMMA fragment loaders (wave32, bf16 K=32) ----------------
// A-matrix 16x32 bf16 (ISA 7.12.2): lane l holds row m=l&15; lanes 0-15 get
// K {0..7,16..23}, lanes 16-31 get K {8..15,24..31} (chunks of 8 halves).
static __device__ __forceinline__ v16bf frag_ld_a(const bf16_t* p) {
  union { v16bf v; uint4 q[2]; } u;
  u.q[0] = *(const uint4*)(p);        // 8 halves at kbase
  u.q[1] = *(const uint4*)(p + 16);   // 8 halves at kbase+16
  return u.v;
}
// B-matrix 32x16 bf16: lane l holds col n=l&15; lanes 0-15 get
// K {0..7,16..23}, lanes 16-31 get K {8..15,24..31} (same as A).
static __device__ __forceinline__ v16bf frag_ld_b(const bf16_t* p) {
  union { v16bf v; uint4 q[2]; } u;
  u.q[0] = *(const uint4*)(p);
  u.q[1] = *(const uint4*)(p + 16);
  return u.v;
}

// ---------------- generic NT GEMM: C[M,N] = A[M,K] * W[N,K]^T ----------------
// EPI bits: 1=+bias[n]  2=ReLU  4=store f32  8=store bf16
// MASK: 0 none, 1 pad-mask(tokens[n]!=0), 2 causal(n<=m)+pad-mask. scale applied pre-mask.
constexpr int TM = 128, TN = 128, TK = 32, LP = 40;   // LDS row stride (halves)

template <bool WF32, int EPI, int MASK>
__global__ __launch_bounds__(256) void gemm_nt(
    const bf16_t* __restrict__ A, int lda, long long a_o, long long a_i,
    const void* __restrict__ Wv, int ldw, long long w_o, long long w_i,
    float* __restrict__ oF, bf16_t* __restrict__ oH, int ldo,
    long long o_o, long long o_i,
    const float* __restrict__ bias, const int* __restrict__ toks, int tokld,
    float scale, int Mdim, int Ndim, int Kdim, int zdiv) {
  __shared__ __align__(16) bf16_t sA[2][TM * LP];
  __shared__ __align__(16) bf16_t sB[2][TN * LP];

  const int z  = blockIdx.z;
  const int zo = z / zdiv, zi = z - zo * zdiv;
  const long long aoff = (long long)zo * a_o + (long long)zi * a_i;
  const long long woff = (long long)zo * w_o + (long long)zi * w_i;
  const long long ooff = (long long)zo * o_o + (long long)zi * o_i;

  const int tn0 = blockIdx.x * TN, tm0 = blockIdx.y * TM;
  const int t = threadIdx.x;
  const int srow = t >> 1, scol = (t & 1) << 4;   // 2 threads/row, 16 elems each
  int ar = tm0 + srow; if (ar >= Mdim) ar = Mdim - 1;
  int wr = tn0 + srow; if (wr >= Ndim) wr = Ndim - 1;

  const int wid = t >> 5, lane = t & 31;
  const int wm = (wid >> 1) * 32, wn = (wid & 1) * 64;
  const int lm = lane & 15, lhA = (lane >> 4) << 3, lhB = (lane >> 4) << 3;

  v8f acc[2][4];
#pragma unroll
  for (int i = 0; i < 2; ++i)
#pragma unroll
    for (int j = 0; j < 4; ++j) acc[i][j] = v8f{0.f,0.f,0.f,0.f,0.f,0.f,0.f,0.f};

  const bf16_t* Ap = A + aoff + (long long)ar * lda + scol;

  // stage tile k0 into LDS buffer `buf` (A always async; W async when bf16)
  auto stage = [&](int buf, int k0) {
    bf16_t* la = &sA[buf][srow * LP + scol];
    const bf16_t* ga = Ap + k0;
    async_cp_b128(la, ga);
    async_cp_b128(la + 8, ga + 8);
    if constexpr (WF32) {  // f32 weights: sync load + convert + ds_store
      const float* Wp = (const float*)Wv + woff + (long long)wr * ldw + k0 + scol;
      float4 f0 = ((const float4*)Wp)[0], f1 = ((const float4*)Wp)[1];
      float4 f2 = ((const float4*)Wp)[2], f3 = ((const float4*)Wp)[3];
      union { uint4 q[2]; bf16_t h[16]; } u;
      u.h[0]=(bf16_t)f0.x; u.h[1]=(bf16_t)f0.y; u.h[2]=(bf16_t)f0.z; u.h[3]=(bf16_t)f0.w;
      u.h[4]=(bf16_t)f1.x; u.h[5]=(bf16_t)f1.y; u.h[6]=(bf16_t)f1.z; u.h[7]=(bf16_t)f1.w;
      u.h[8]=(bf16_t)f2.x; u.h[9]=(bf16_t)f2.y; u.h[10]=(bf16_t)f2.z; u.h[11]=(bf16_t)f2.w;
      u.h[12]=(bf16_t)f3.x; u.h[13]=(bf16_t)f3.y; u.h[14]=(bf16_t)f3.z; u.h[15]=(bf16_t)f3.w;
      *(uint4*)&sB[buf][srow * LP + scol]     = u.q[0];
      *(uint4*)&sB[buf][srow * LP + scol + 8] = u.q[1];
    } else {
      bf16_t* lb = &sB[buf][srow * LP + scol];
      const bf16_t* gw = (const bf16_t*)Wv + woff + (long long)wr * ldw + k0 + scol;
      async_cp_b128(lb, gw);
      async_cp_b128(lb + 8, gw + 8);
    }
  };

  constexpr int NPER = WF32 ? 2 : 4;   // async instructions issued per wave per stage
  const int nsteps = Kdim / TK;

  stage(0, 0);
  for (int s = 0; s < nsteps; ++s) {
    const int cur = s & 1;
    const bool more = (s + 1 < nsteps);
    if (more) stage(1 - cur, (s + 1) * TK);   // prefetch next tile into other buffer
    if (more) wait_asyncN<NPER>();            // newest NPER outstanding = next tile
    else      wait_asyncN<0>();
    __syncthreads();                          // cur tile visible to all waves

    v16bf fa[2], fb[4];
#pragma unroll
    for (int mi = 0; mi < 2; ++mi)
      fa[mi] = frag_ld_a(&sA[cur][(wm + mi * 16 + lm) * LP + lhA]);
#pragma unroll
    for (int ni = 0; ni < 4; ++ni)
      fb[ni] = frag_ld_b(&sB[cur][(wn + ni * 16 + lm) * LP + lhB]);
#pragma unroll
    for (int mi = 0; mi < 2; ++mi)
#pragma unroll
      for (int ni = 0; ni < 4; ++ni)
        acc[mi][ni] = __builtin_amdgcn_wmma_f32_16x16x32_bf16(
            false, fa[mi], false, fb[ni], (short)0, acc[mi][ni], false, false);
    __syncthreads();                          // all reads of cur done before overwrite
  }

  // epilogue: C/D layout -> lane n = lane&15, row m = r + 8*(lane>>4)
  const int lh8 = (lane >> 4) * 8;
#pragma unroll
  for (int mi = 0; mi < 2; ++mi) {
#pragma unroll
    for (int ni = 0; ni < 4; ++ni) {
      const int n = tn0 + wn + ni * 16 + lm;
      if (n >= Ndim) continue;
      float bv = 0.f;
      if (EPI & 1) bv = bias[n];
      bool tokOK = true;
      if (MASK) tokOK = (toks[zo * tokld + n] != 0);
#pragma unroll
      for (int r = 0; r < 8; ++r) {
        const int m = tm0 + wm + mi * 16 + lh8 + r;
        if (m >= Mdim) continue;
        float v = acc[mi][ni][r] * scale + bv;
        if (EPI & 2) v = v > 0.f ? v : 0.f;
        if (MASK) {
          bool keep = tokOK;
          if (MASK == 2) keep = keep && (n <= m);
          if (!keep) v = -1e30f;
        }
        const long long oi = ooff + (long long)m * ldo + n;
        if (EPI & 4) oF[oi] = v;
        if (EPI & 8) oH[oi] = (bf16_t)v;
      }
    }
  }
}

// ---------------- helper kernels ----------------
__global__ void k_f2h(const float* __restrict__ in, bf16_t* __restrict__ out, long long n) {
  long long i = (long long)blockIdx.x * blockDim.x + threadIdx.x;
  if (i < n) out[i] = (bf16_t)in[i];
}

__global__ void k_embed(const int* __restrict__ toks, const float* __restrict__ emb,
                        float* __restrict__ xf, bf16_t* __restrict__ xh,
                        int rows, int seq) {
  long long idx = (long long)blockIdx.x * blockDim.x + threadIdx.x;
  if (idx >= (long long)rows * D_) return;
  int d = (int)(idx & (D_ - 1));
  int row = (int)(idx >> 9);   // /512
  int pos = row % seq;
  int tk = toks[row];
  float v = emb[(long long)tk * D_ + d] * SQRTD_;
  float freq = __expf((float)(d >> 1) * (-2.0f * 9.210340371976184f / (float)D_));
  float ang = (float)pos * freq;
  v += (d & 1) ? __cosf(ang) : __sinf(ang);
  xf[idx] = v;
  xh[idx] = (bf16_t)v;
}

__global__ __launch_bounds__(256) void k_add_ln(
    const float* __restrict__ x, const float* __restrict__ a,
    const float* __restrict__ g, const float* __restrict__ bb,
    float* __restrict__ oF, bf16_t* __restrict__ oH) {
  __shared__ float r1[256];
  __shared__ float r2[256];
  const int row = blockIdx.x, t = threadIdx.x;
  const long long base = (long long)row * D_;
  float v0 = x[base + t] + a[base + t];
  float v1 = x[base + t + 256] + a[base + t + 256];
  r1[t] = v0 + v1;
  r2[t] = v0 * v0 + v1 * v1;
  __syncthreads();
  for (int s = 128; s > 0; s >>= 1) {
    if (t < s) { r1[t] += r1[t + s]; r2[t] += r2[t + s]; }
    __syncthreads();
  }
  const float mean = r1[0] * (1.0f / D_);
  const float var  = r2[0] * (1.0f / D_) - mean * mean;
  const float rs   = rsqrtf(var + EPS_);
  float o0 = (v0 - mean) * rs * g[t] + bb[t];
  float o1 = (v1 - mean) * rs * g[t + 256] + bb[t + 256];
  oF[base + t] = o0;           oF[base + t + 256] = o1;
  oH[base + t] = (bf16_t)o0;   oH[base + t + 256] = (bf16_t)o1;
}

__global__ __launch_bounds__(256) void k_softmax(const float* __restrict__ sc,
                                                 bf16_t* __restrict__ pr) {
  __shared__ float red[256];
  const int row = blockIdx.x, t = threadIdx.x;
  const long long base = (long long)row * S_;
  float v0 = sc[base + t], v1 = sc[base + t + 256];
  red[t] = fmaxf(v0, v1);
  __syncthreads();
  for (int s = 128; s > 0; s >>= 1) {
    if (t < s) red[t] = fmaxf(red[t], red[t + s]);
    __syncthreads();
  }
  const float mx = red[0];
  __syncthreads();
  float e0 = __expf(v0 - mx), e1 = __expf(v1 - mx);
  red[t] = e0 + e1;
  __syncthreads();
  for (int s = 128; s > 0; s >>= 1) {
    if (t < s) red[t] += red[t + s];
    __syncthreads();
  }
  const float inv = 1.0f / red[0];
  pr[base + t]       = (bf16_t)(e0 * inv);
  pr[base + t + 256] = (bf16_t)(e1 * inv);
}

// vt[b][h][d][s] = v[b*S+s][h*HD+d]
__global__ void k_transpose_v(const bf16_t* __restrict__ v, bf16_t* __restrict__ vt) {
  long long idx = (long long)blockIdx.x * blockDim.x + threadIdx.x;
  if (idx >= (long long)B_ * H_ * HD_ * S_) return;
  int s = (int)(idx % S_);
  long long r = idx / S_;
  int d = (int)(r % HD_); r /= HD_;
  int h = (int)(r % H_);
  int b = (int)(r / H_);
  vt[idx] = v[((long long)(b * S_ + s)) * D_ + h * HD_ + d];
}

// ---------------- host orchestration ----------------
extern "C" void kernel_launch(void* const* d_in, const int* in_sizes, int n_in,
                              void* d_out, int out_size, void* d_ws, size_t ws_size,
                              hipStream_t stream) {
  (void)in_sizes; (void)n_in; (void)out_size; (void)ws_size;
  const int*   src   = (const int*)d_in[0];
  const int*   tgt   = (const int*)d_in[1];
  const float* emb   = (const float*)d_in[2];
  const float* e_wq  = (const float*)d_in[3];
  const float* e_wk  = (const float*)d_in[4];
  const float* e_wv  = (const float*)d_in[5];
  const float* e_wo  = (const float*)d_in[6];
  const float* e_g1  = (const float*)d_in[7];
  const float* e_b1  = (const float*)d_in[8];
  const float* e_g2  = (const float*)d_in[9];
  const float* e_b2  = (const float*)d_in[10];
  const float* e_f1w = (const float*)d_in[11];
  const float* e_f1b = (const float*)d_in[12];
  const float* e_f2w = (const float*)d_in[13];
  const float* e_f2b = (const float*)d_in[14];
  const float* d_swq = (const float*)d_in[15];
  const float* d_swk = (const float*)d_in[16];
  const float* d_swv = (const float*)d_in[17];
  const float* d_swo = (const float*)d_in[18];
  const float* d_cwq = (const float*)d_in[19];
  const float* d_cwk = (const float*)d_in[20];
  const float* d_cwv = (const float*)d_in[21];
  const float* d_cwo = (const float*)d_in[22];
  const float* d_g1  = (const float*)d_in[23];
  const float* d_b1  = (const float*)d_in[24];
  const float* d_g2  = (const float*)d_in[25];
  const float* d_b2  = (const float*)d_in[26];
  const float* d_g3  = (const float*)d_in[27];
  const float* d_b3  = (const float*)d_in[28];
  const float* d_f1w = (const float*)d_in[29];
  const float* d_f1b = (const float*)d_in[30];
  const float* d_f2w = (const float*)d_in[31];
  const float* d_f2b = (const float*)d_in[32];

  // workspace carve (256B aligned)
  char* wp = (char*)d_ws;
  auto take = [&](size_t bytes) -> void* {
    void* p = wp; wp += (bytes + 255) & ~(size_t)255; return p;
  };
  float*  xF   = (float*)take((size_t)M_ * D_ * 4);
  bf16_t* xH   = (bf16_t*)take((size_t)M_ * D_ * 2);
  float*  yF   = (float*)take((size_t)M_ * D_ * 4);
  bf16_t* yH   = (bf16_t*)take((size_t)M_ * D_ * 2);
  bf16_t* qH   = (bf16_t*)take((size_t)M_ * D_ * 2);
  bf16_t* kH   = (bf16_t*)take((size_t)M_ * D_ * 2);
  bf16_t* vH   = (bf16_t*)take((size_t)M_ * D_ * 2);
  bf16_t* vtH  = (bf16_t*)take((size_t)M_ * D_ * 2);
  bf16_t* ctxH = (bf16_t*)take((size_t)M_ * D_ * 2);
  float*  scF  = (float*)take((size_t)B_ * H_ * S_ * S_ * 4);
  bf16_t* prH  = (bf16_t*)take((size_t)B_ * H_ * S_ * S_ * 2);
  float*  atF  = (float*)take((size_t)M_ * D_ * 4);
  bf16_t* ffH  = (bf16_t*)take((size_t)M_ * DFF_ * 2);
  float*  ffF  = (float*)take((size_t)M_ * D_ * 4);
  bf16_t* embH = (bf16_t*)take((size_t)V_ * D_ * 2);

  const dim3 blk(256);
  auto grd = [](int N, int M, int Z) { return dim3((N + TN - 1) / TN, (M + TM - 1) / TM, Z); };

  // projections / FFN (fp32 weights converted in-kernel)
  auto projB = [&](const bf16_t* Ain, const float* Wm, bf16_t* outH, int Mm, int Nn, int Kk) {
    gemm_nt<true, 8, 0><<<grd(Nn, Mm, 1), blk, 0, stream>>>(
        Ain, Kk, 0, 0, Wm, Kk, 0, 0, nullptr, outH, Nn, 0, 0,
        nullptr, nullptr, 0, 1.0f, Mm, Nn, Kk, 1);
  };
  auto projF = [&](const bf16_t* Ain, const float* Wm, float* outF, int Mm, int Nn, int Kk) {
    gemm_nt<true, 4, 0><<<grd(Nn, Mm, 1), blk, 0, stream>>>(
        Ain, Kk, 0, 0, Wm, Kk, 0, 0, outF, nullptr, Nn, 0, 0,
        nullptr, nullptr, 0, 1.0f, Mm, Nn, Kk, 1);
  };
  auto ffn = [&](const bf16_t* xin, const float* w1, const float* b1,
                 const float* w2, const float* b2, float* outF) {
    gemm_nt<true, 1 | 2 | 8, 0><<<grd(DFF_, M_, 1), blk, 0, stream>>>(
        xin, D_, 0, 0, w1, D_, 0, 0, nullptr, ffH, DFF_, 0, 0,
        b1, nullptr, 0, 1.0f, M_, DFF_, D_, 1);
    gemm_nt<true, 1 | 4, 0><<<grd(D_, M_, 1), blk, 0, stream>>>(
        ffH, DFF_, 0, 0, w2, DFF_, 0, 0, outF, nullptr, D_, 0, 0,
        b2, nullptr, 0, 1.0f, M_, D_, DFF_, 1);
  };
  auto addln = [&](float* xio, bf16_t* xh, const float* aa, const float* g, const float* b) {
    k_add_ln<<<M_, blk, 0, stream>>>(xio, aa, g, b, xio, xh);
  };

  // full attention block: out = (softmax(mask(Q K^T * scale)) V) Wo^T
  auto attention = [&](const bf16_t* xq, const bf16_t* xkv,
                       const float* wq, const float* wk, const float* wv, const float* wo,
                       const int* mtoks, int causal, float* outAt) {
    projB(xq,  wq, qH, M_, D_, D_);
    projB(xkv, wk, kH, M_, D_, D_);
    projB(xkv, wv, vH, M_, D_, D_);
    k_transpose_v<<<(B_ * H_ * HD_ * S_ + 255) / 256, blk, 0, stream>>>(vH, vtH);
    const long long aO = (long long)S_ * D_, aI = HD_;
    const long long sO = (long long)H_ * S_ * S_, sI = (long long)S_ * S_;
    if (causal)
      gemm_nt<false, 4, 2><<<grd(S_, S_, B_ * H_), blk, 0, stream>>>(
          qH, D_, aO, aI, kH, D_, aO, aI, scF, nullptr, S_, sO, sI,
          nullptr, mtoks, S_, SCALE_, S_, S_, HD_, H_);
    else
      gemm_nt<false, 4, 1><<<grd(S_, S_, B_ * H_), blk, 0, stream>>>(
          qH, D_, aO, aI, kH, D_, aO, aI, scF, nullptr, S_, sO, sI,
          nullptr, mtoks, S_, SCALE_, S_, S_, HD_, H_);
    k_softmax<<<B_ * H_ * S_, blk, 0, stream>>>(scF, prH);
    gemm_nt<false, 8, 0><<<grd(HD_, S_, B_ * H_), blk, 0, stream>>>(
        prH, S_, sO, sI, vtH, S_, (long long)H_ * HD_ * S_, (long long)HD_ * S_,
        nullptr, ctxH, D_, (long long)S_ * D_, (long long)HD_,
        nullptr, nullptr, 0, 1.0f, S_, HD_, S_, H_);
    projF(ctxH, wo, outAt, M_, D_, D_);
  };

  // ---- tied embedding table in bf16 (for logits GEMM) ----
  k_f2h<<<(int)(((long long)V_ * D_ + 255) / 256), blk, 0, stream>>>(emb, embH, (long long)V_ * D_);

  // ---- encoder ----
  k_embed<<<(M_ * D_ + 255) / 256, blk, 0, stream>>>(src, emb, xF, xH, M_, S_);
  for (int i = 0; i < LE_; ++i) {
    const size_t dd = (size_t)i * D_ * D_;
    attention(xH, xH, e_wq + dd, e_wk + dd, e_wv + dd, e_wo + dd, src, 0, atF);
    addln(xF, xH, atF, e_g1 + i * D_, e_b1 + i * D_);
    ffn(xH, e_f1w + (size_t)i * DFF_ * D_, e_f1b + (size_t)i * DFF_,
        e_f2w + (size_t)i * D_ * DFF_, e_f2b + (size_t)i * D_, ffF);
    addln(xF, xH, ffF, e_g2 + i * D_, e_b2 + i * D_);
  }
  // xH now holds encoder memory (bf16)

  // ---- decoder ----
  k_embed<<<(M_ * D_ + 255) / 256, blk, 0, stream>>>(tgt, emb, yF, yH, M_, T_);
  for (int i = 0; i < LD_; ++i) {
    const size_t dd = (size_t)i * D_ * D_;
    attention(yH, yH, d_swq + dd, d_swk + dd, d_swv + dd, d_swo + dd, tgt, 1, atF);
    addln(yF, yH, atF, d_g1 + i * D_, d_b1 + i * D_);
    attention(yH, xH, d_cwq + dd, d_cwk + dd, d_cwv + dd, d_cwo + dd, src, 0, atF);
    addln(yF, yH, atF, d_g2 + i * D_, d_b2 + i * D_);
    ffn(yH, d_f1w + (size_t)i * DFF_ * D_, d_f1b + (size_t)i * DFF_,
        d_f2w + (size_t)i * D_ * DFF_, d_f2b + (size_t)i * D_, ffF);
    addln(yF, yH, ffF, d_g3 + i * D_, d_b3 + i * D_);
  }

  // ---- logits: [2048,512] x [32000,512]^T -> d_out f32 ----
  gemm_nt<false, 4, 0><<<grd(V_, M_, 1), blk, 0, stream>>>(
      yH, D_, 0, 0, embH, D_, 0, 0, (float*)d_out, nullptr, V_, 0, 0,
      nullptr, nullptr, 0, 1.0f, M_, V_, D_, 1);
}
